// TokenFilter_33560874451582
// MI455X (gfx1250) — matrix operand order repaired and checked by reading, not compiled, not run on hardware
//
#include <hip/hip_runtime.h>
#include <hip/hip_bf16.h>

typedef __attribute__((ext_vector_type(16))) _Float16 v16h;
typedef __attribute__((ext_vector_type(8)))  float    v8f;
typedef __attribute__((ext_vector_type(4)))  int      v4i;

// ---------------------------------------------------------------------------
// CDNA5 async global->LDS path (ASYNCcnt), guarded so compile never breaks.
// Builtin signature (leaked by clang diagnostic): params are pointers to
// int4 in the global (AS1) / LDS (AS3) address spaces.
// ---------------------------------------------------------------------------
#if defined(__has_builtin)
#if __has_builtin(__builtin_amdgcn_global_load_async_to_lds_b128)
#define HAVE_ASYNC_LDS 1
#endif
#endif
#ifndef HAVE_ASYNC_LDS
#define HAVE_ASYNC_LDS 0
#endif

// copy 16 contiguous bytes of f16 data from global to LDS
__device__ __forceinline__ void copy16_g2l(const _Float16* g, _Float16* l) {
#if HAVE_ASYNC_LDS
  typedef __attribute__((address_space(1))) v4i as1_v4i;
  typedef __attribute__((address_space(3))) v4i as3_v4i;
  // reinterpret through integers: flat global address == AS1 address;
  // low 32 bits of a flat LDS address == AS3 byte offset (AS3 ptrs are 32-bit)
  as1_v4i* gp = (as1_v4i*)(unsigned long long)(const void*)g;
  as3_v4i* lp = (as3_v4i*)(unsigned)(unsigned long long)(void*)l;
  __builtin_amdgcn_global_load_async_to_lds_b128(gp, lp, 0, 0);
#else
  *(uint4*)l = *(const uint4*)g;
#endif
}

template <int NN>
__device__ __forceinline__ void wait_async() {
#if HAVE_ASYNC_LDS
#if __has_builtin(__builtin_amdgcn_s_wait_asynccnt)
  __builtin_amdgcn_s_wait_asynccnt(NN);
#else
  asm volatile("s_wait_asynccnt %0" ::"i"(NN) : "memory");
#endif
#endif
}

// ---------------------------------------------------------------------------
// WMMA helpers (CDNA5 v_wmma_f32_16x16x32_f16, wave32)
// ---------------------------------------------------------------------------
__device__ __forceinline__ v8f wmma_f16(v16h a, v16h b, v8f c) {
  return __builtin_amdgcn_wmma_f32_16x16x32_f16(
      false, a, false, b, (short)0, c, false, false);
}

// A fragment: 16x32 f16 tile, row-major in LDS with given stride.
__device__ __forceinline__ v16h load_frag_a(const _Float16* base, int stride, int lane) {
  int row  = lane & 15;
  int koff = (lane & 16) ? 8 : 0;
  const _Float16* p = base + row * stride + koff;
  v16h a;
#pragma unroll
  for (int i = 0; i < 8; ++i) { a[i] = p[i]; a[i + 8] = p[16 + i]; }
  return a;
}

// B fragment: logical 32x16 (KxN); LDS holds Bt[n][k] row-major with stride.
__device__ __forceinline__ v16h load_frag_b(const _Float16* baseT, int stride, int lane) {
  int col  = lane & 15;
  int koff = (lane & 16) ? 8 : 0;
  const _Float16* p = baseT + col * stride + koff;
  v16h b;
#pragma unroll
  for (int i = 0; i < 8; ++i) { b[i] = p[i]; b[i + 8] = p[16 + i]; }
  return b;
}

// ---------------------------------------------------------------------------
// 1) xp_f16 = (f16)(x + pos_emb)
// ---------------------------------------------------------------------------
__global__ __launch_bounds__(256) void add_pos_kernel(
    const float* __restrict__ x, const float* __restrict__ pe,
    _Float16* __restrict__ xph, int total, int nd) {
  int i = blockIdx.x * 256 + threadIdx.x;
  if (i < total) xph[i] = (_Float16)(x[i] + pe[i % nd]);
}

// f32 in[R,C] -> f16 out[C,R] (transposed), tiles 32x32, dims % 32 == 0
__global__ __launch_bounds__(256) void transpose_h_kernel(
    const float* __restrict__ in, _Float16* __restrict__ out, int R, int C) {
  __shared__ float t[32][33];
  const int bx = blockIdx.x * 32, by = blockIdx.y * 32;
  const int tx = threadIdx.x & 31, ty = threadIdx.x >> 5;
  for (int i = ty; i < 32; i += 8)
    t[i][tx] = in[(long long)(by + i) * C + bx + tx];
  __syncthreads();
  for (int i = ty; i < 32; i += 8)
    out[(long long)(bx + i) * R + by + tx] = (_Float16)t[tx][i];
}

// ---------------------------------------------------------------------------
// 2) GEMM: C[M,N] = A[M,K](f16) * Bt[N,K]^T(f16) + bias (+ residual f16)
//    Block: 256 thr = 8 waves, 128x64 tile; wave (wm,wn) does 32x32 (4 WMMA).
//    Double-buffered LDS, async b128 staging (3 copy16 per thread per tile).
//    Requires M%128==0, N%64==0, K%32==0 (true at all call sites).
// ---------------------------------------------------------------------------
__global__ __launch_bounds__(256) void gemm_f16_kernel(
    const _Float16* __restrict__ A, const _Float16* __restrict__ Bt,
    const float* __restrict__ bias, const _Float16* __restrict__ residual,
    void* __restrict__ Cout, int c_is_f32,
    int M, int N, int K, long long sA, long long sB, long long sC) {
  __shared__ __align__(16) _Float16 ldsA[2][128 * 32];
  __shared__ __align__(16) _Float16 ldsB[2][64 * 32];
  const int tid = threadIdx.x, lane = tid & 31, wid = tid >> 5;
  const int wm = wid & 3, wn = wid >> 2;
  const int gm = blockIdx.y * 128, gn = blockIdx.x * 64;
  const _Float16* Ab = A + (long long)blockIdx.z * sA;
  const _Float16* Bb = Bt + (long long)blockIdx.z * sB;

  // staging geometry: A = 128 rows x 4 eight-half chunks (2 per thread);
  //                   B = 64 rows x 4 chunks (1 per thread)
  const int ar = tid >> 2, kc = (tid & 3) * 8;

  v8f acc00 = {}, acc01 = {}, acc10 = {}, acc11 = {};
  int cur = 0;

#define GEMM_ISSUE(K0, BUF)                                                     \
  do {                                                                          \
    copy16_g2l(Ab + (long long)(gm + ar) * K + (K0) + kc,                       \
               &ldsA[BUF][ar * 32 + kc]);                                       \
    copy16_g2l(Ab + (long long)(gm + 64 + ar) * K + (K0) + kc,                  \
               &ldsA[BUF][(64 + ar) * 32 + kc]);                                \
    copy16_g2l(Bb + (long long)(gn + ar) * K + (K0) + kc,                       \
               &ldsB[BUF][ar * 32 + kc]);                                       \
  } while (0)

  GEMM_ISSUE(0, 0);
  for (int k0 = 0; k0 < K; k0 += 32) {
    const bool more = (k0 + 32) < K;
    if (more) {
      GEMM_ISSUE(k0 + 32, cur ^ 1);
      wait_async<3>();
    } else {
      wait_async<0>();
    }
    __syncthreads();
    const _Float16* bA = ldsA[cur];
    const _Float16* bB = ldsB[cur];
    v16h a0 = load_frag_a(bA + (wm * 32) * 32, 32, lane);
    v16h a1 = load_frag_a(bA + (wm * 32 + 16) * 32, 32, lane);
    v16h b0 = load_frag_b(bB + (wn * 32) * 32, 32, lane);
    v16h b1 = load_frag_b(bB + (wn * 32 + 16) * 32, 32, lane);
    acc00 = wmma_f16(a0, b0, acc00);
    acc01 = wmma_f16(a0, b1, acc01);
    acc10 = wmma_f16(a1, b0, acc10);
    acc11 = wmma_f16(a1, b1, acc11);
    __syncthreads();
    cur ^= 1;
  }
#undef GEMM_ISSUE

  const int col = lane & 15, rb = (lane & 16) ? 8 : 0;
  float*    Cf = (float*)Cout + (long long)blockIdx.z * sC;
  _Float16* Ch = (_Float16*)Cout + (long long)blockIdx.z * sC;
  const _Float16* Rb = residual ? residual + (long long)blockIdx.z * sC : nullptr;
#pragma unroll
  for (int i = 0; i < 2; ++i) {
#pragma unroll
    for (int t = 0; t < 2; ++t) {
      v8f acc = (i == 0) ? (t == 0 ? acc00 : acc01) : (t == 0 ? acc10 : acc11);
      int n = gn + wn * 32 + t * 16 + col;
      float bv = bias ? bias[n] : 0.0f;
#pragma unroll
      for (int r = 0; r < 8; ++r) {
        long long off = (long long)(gm + wm * 32 + i * 16 + r + rb) * N + n;
        float v = acc[r] + bv;
        if (Rb) v += (float)Rb[off];
        if (c_is_f32) Cf[off] = v;
        else          Ch[off] = (_Float16)v;
      }
    }
  }
}

// ---------------------------------------------------------------------------
// 3) importance[b,i] = 1 - 1/sum_j exp(sim[b,i,j] - max_j)   (diag = -inf)
// ---------------------------------------------------------------------------
__global__ __launch_bounds__(256) void importance_kernel(
    const float* __restrict__ sim, float* __restrict__ imp) {
  __shared__ float red[256];
  const int bi = blockIdx.x;
  const int i = bi & 255, tid = threadIdx.x;
  float v = sim[(long long)bi * 256 + tid];
  if (tid == i) v = -3.4e38f;
  red[tid] = v;
  __syncthreads();
  for (int s = 128; s; s >>= 1) {
    if (tid < s) red[tid] = fmaxf(red[tid], red[tid + s]);
    __syncthreads();
  }
  float mx = red[0];
  __syncthreads();
  red[tid] = __expf(v - mx);
  __syncthreads();
  for (int s = 128; s; s >>= 1) {
    if (tid < s) red[tid] += red[tid + s];
    __syncthreads();
  }
  if (tid == 0) imp[bi] = 1.0f - 1.0f / red[0];
}

// ---------------------------------------------------------------------------
// 4) top-64 per batch (descending, ties -> lower index), then gather rows
// ---------------------------------------------------------------------------
__global__ __launch_bounds__(256) void topk_kernel(
    const float* __restrict__ imp, int* __restrict__ idx) {
  __shared__ float vals[256];
  __shared__ float rv[256];
  __shared__ int   ri[256];
  const int b = blockIdx.x, tid = threadIdx.x;
  vals[tid] = imp[b * 256 + tid];
  __syncthreads();
  for (int s = 0; s < 64; ++s) {
    rv[tid] = vals[tid]; ri[tid] = tid;
    __syncthreads();
    for (int st = 128; st; st >>= 1) {
      if (tid < st) {
        float ov = rv[tid + st]; int oi = ri[tid + st];
        if (ov > rv[tid] || (ov == rv[tid] && oi < ri[tid])) { rv[tid] = ov; ri[tid] = oi; }
      }
      __syncthreads();
    }
    if (tid == 0) { idx[b * 64 + s] = ri[0]; vals[ri[0]] = -3.4e38f; }
    __syncthreads();
  }
}

__global__ __launch_bounds__(256) void gather_kernel(
    const _Float16* __restrict__ xph, const int* __restrict__ idx,
    _Float16* __restrict__ sel) {
  const int bs = blockIdx.x;
  const int b = bs >> 6;
  const int src = idx[bs];
  const unsigned* in  = (const unsigned*)(xph + ((long long)(b * 256 + src)) * 1024);
  unsigned*       out = (unsigned*)(sel + (long long)bs * 1024);
  for (int e = threadIdx.x; e < 512; e += 256) out[e] = in[e];
}

// ---------------------------------------------------------------------------
// 5) Fused attention per (b,h): q(64x32) k(256x32) v(256x32), f16 in, f16 out.
//    128 thr = 4 waves; wave w owns query rows [16w,16w+16).
//    1/sqrt(dh) folded into the softmax exponent so Q/K staging is a raw copy.
//    Dynamic LDS 112KB, phase-overlapped regions.
// ---------------------------------------------------------------------------
__global__ __launch_bounds__(128) void attn_kernel(
    const _Float16* __restrict__ Q, const _Float16* __restrict__ Kp,
    const _Float16* __restrict__ Vp, _Float16* __restrict__ O) {
  extern __shared__ char smem[];
  float*    sS  = (float*)smem;                           // 65536 B scores
  _Float16* sP  = (_Float16*)(smem + 65536);              // 32768 B probs
  _Float16* sVt = (_Float16*)(smem + 65536 + 32768);      // 16384 B V^T
  _Float16* sQ  = (_Float16*)(smem + 65536);              // 4096 B (dead before sP)
  _Float16* sK  = (_Float16*)(smem + 65536 + 4096);       // 16384 B (dead before sP)

  const int b = blockIdx.x >> 5, h = blockIdx.x & 31;
  const int tid = threadIdx.x, lane = tid & 31, w = tid >> 5;
  const float qscale = 0.17677669529663687f;  // 1/sqrt(32)

  // async raw copies: Q 2 chunks/thread, K 8 chunks/thread
  for (int c = tid; c < 256; c += 128) {
    int r = c >> 2, kc2 = (c & 3) * 8;
    copy16_g2l(Q + (long long)(b * 64 + r) * 1024 + h * 32 + kc2, &sQ[r * 32 + kc2]);
  }
  for (int c = tid; c < 1024; c += 128) {
    int s = c >> 2, kc2 = (c & 3) * 8;
    copy16_g2l(Kp + (long long)(b * 256 + s) * 1024 + h * 32 + kc2, &sK[s * 32 + kc2]);
  }
  // V^T staging: vector global load, scalar LDS scatter
  for (int c = tid; c < 1024; c += 128) {
    int s = c >> 2, dc = (c & 3) * 8;
    const _Float16* src = Vp + (long long)(b * 256 + s) * 1024 + h * 32 + dc;
    _Float16 tmp[8];
    *(uint4*)tmp = *(const uint4*)src;
#pragma unroll
    for (int j = 0; j < 8; ++j) sVt[(dc + j) * 256 + s] = tmp[j];
  }
  wait_async<0>();
  __syncthreads();

  const int col = lane & 15, rb = (lane & 16) ? 8 : 0;
  {
    v16h af = load_frag_a(sQ + (w * 16) * 32, 32, lane);
    for (int nt = 0; nt < 16; ++nt) {
      v8f acc = {};
      v16h bf = load_frag_b(sK + (nt * 16) * 32, 32, lane);
      acc = wmma_f16(af, bf, acc);
#pragma unroll
      for (int r = 0; r < 8; ++r)
        sS[(w * 16 + r + rb) * 256 + nt * 16 + col] = acc[r];
    }
  }
  __syncthreads();

  if (tid < 64) {
    const float* row = sS + tid * 256;
    float mx = -3.4e38f;
    for (int j = 0; j < 256; ++j) mx = fmaxf(mx, row[j]);
    float sum = 0.0f;
    for (int j = 0; j < 256; ++j) sum += __expf((row[j] - mx) * qscale);
    float inv = 1.0f / sum;
    _Float16* prow = sP + tid * 256;
    for (int j = 0; j < 256; ++j)
      prow[j] = (_Float16)(__expf((row[j] - mx) * qscale) * inv);
  }
  __syncthreads();

  v8f o0 = {}, o1 = {};
  for (int k0 = 0; k0 < 256; k0 += 32) {
    v16h pa = load_frag_a(sP + (w * 16) * 256 + k0, 256, lane);
    v16h b0 = load_frag_b(sVt + 0 * 256 + k0, 256, lane);
    o0 = wmma_f16(pa, b0, o0);
    v16h b1 = load_frag_b(sVt + 16 * 256 + k0, 256, lane);
    o1 = wmma_f16(pa, b1, o1);
  }
#pragma unroll
  for (int t = 0; t < 2; ++t) {
    v8f acc = t ? o1 : o0;
#pragma unroll
    for (int r = 0; r < 8; ++r)
      O[((long long)(b * 64 + w * 16 + r + rb)) * 1024 + h * 32 + t * 16 + col] =
          (_Float16)acc[r];
  }
}

// ---------------------------------------------------------------------------
// Host side
// ---------------------------------------------------------------------------
extern "C" void kernel_launch(void* const* d_in, const int* in_sizes, int n_in,
                              void* d_out, int out_size, void* d_ws, size_t ws_size,
                              hipStream_t stream) {
  const int B = 128, N = 256, D = 1024, DO = 4096, KSEL = 64;
  const float* x  = (const float*)d_in[0];
  const float* pe = (const float*)d_in[1];
  const float* wq = (const float*)d_in[2];
  const float* bq = (const float*)d_in[3];
  const float* wk = (const float*)d_in[4];
  const float* bk = (const float*)d_in[5];
  const float* wv = (const float*)d_in[6];
  const float* bv = (const float*)d_in[7];
  const float* wd = (const float*)d_in[8];
  const float* bd = (const float*)d_in[9];
  const float* wp = (const float*)d_in[10];
  const float* bp = (const float*)d_in[11];
  float* out = (float*)d_out;

  // bump allocator over d_ws (256B aligned regions)
  char* wsp = (char*)d_ws;
  auto alloc = [&](long long bytes) {
    char* p = wsp;
    wsp += (bytes + 255) & ~255LL;
    return (void*)p;
  };
  _Float16* xph = (_Float16*)alloc((long long)B * N * D * 2);     // f16 x+pos
  _Float16* wTq = (_Float16*)alloc((long long)D * D * 2);
  _Float16* wTk = (_Float16*)alloc((long long)D * D * 2);
  _Float16* wTv = (_Float16*)alloc((long long)D * D * 2);
  _Float16* wTd = (_Float16*)alloc((long long)D * D * 2);
  _Float16* wTp = (_Float16*)alloc((long long)D * DO * 2);
  float*    sim = (float*)alloc((long long)B * N * N * 4);
  float*    imp = (float*)alloc((long long)B * N * 4);
  int*      tk  = (int*)alloc((long long)B * KSEL * 4);
  _Float16* sel = (_Float16*)alloc((long long)B * KSEL * D * 2);
  _Float16* Qb  = (_Float16*)alloc((long long)B * KSEL * D * 2);
  _Float16* Kb  = (_Float16*)alloc((long long)B * N * D * 2);
  _Float16* Vb  = (_Float16*)alloc((long long)B * N * D * 2);
  _Float16* AO  = (_Float16*)alloc((long long)B * KSEL * D * 2);
  _Float16* PO  = (_Float16*)alloc((long long)B * KSEL * D * 2);

  const int totalX = B * N * D;

  // prep: f16 conversions + weight transposes
  add_pos_kernel<<<(totalX + 255) / 256, 256, 0, stream>>>(x, pe, xph, totalX, N * D);
  transpose_h_kernel<<<dim3(D / 32, D / 32), 256, 0, stream>>>(wq, wTq, D, D);
  transpose_h_kernel<<<dim3(D / 32, D / 32), 256, 0, stream>>>(wk, wTk, D, D);
  transpose_h_kernel<<<dim3(D / 32, D / 32), 256, 0, stream>>>(wv, wTv, D, D);
  transpose_h_kernel<<<dim3(D / 32, D / 32), 256, 0, stream>>>(wd, wTd, D, D);
  transpose_h_kernel<<<dim3(DO / 32, D / 32), 256, 0, stream>>>(wp, wTp, D, DO);

  // sim = xp @ xp^T  (batched; Bt == xph rows, already [N][K])
  gemm_f16_kernel<<<dim3(N / 64, N / 128, B), 256, 0, stream>>>(
      xph, xph, nullptr, nullptr, sim, 1, N, N, D,
      (long long)N * D, (long long)N * D, (long long)N * N);

  importance_kernel<<<B * N, 256, 0, stream>>>(sim, imp);
  topk_kernel<<<B, 256, 0, stream>>>(imp, tk);
  gather_kernel<<<B * KSEL, 256, 0, stream>>>(xph, tk, sel);

  // projections (f16 out)
  gemm_f16_kernel<<<dim3(D / 64, (B * KSEL) / 128, 1), 256, 0, stream>>>(
      sel, wTq, bq, nullptr, Qb, 0, B * KSEL, D, D, 0, 0, 0);
  gemm_f16_kernel<<<dim3(D / 64, (B * N) / 128, 1), 256, 0, stream>>>(
      xph, wTk, bk, nullptr, Kb, 0, B * N, D, D, 0, 0, 0);
  gemm_f16_kernel<<<dim3(D / 64, (B * N) / 128, 1), 256, 0, stream>>>(
      xph, wTv, bv, nullptr, Vb, 0, B * N, D, D, 0, 0, 0);

  attn_kernel<<<B * 32, 128, 114688, stream>>>(Qb, Kb, Vb, AO);

  // pooled = AO@wd + bd + sel (f16 out)
  gemm_f16_kernel<<<dim3(D / 64, (B * KSEL) / 128, 1), 256, 0, stream>>>(
      AO, wTd, bd, sel, PO, 0, B * KSEL, D, D, 0, 0, 0);

  // out = pooled@wp + bp (f32 out)
  gemm_f16_kernel<<<dim3(DO / 64, (B * KSEL) / 128, 1), 256, 0, stream>>>(
      PO, wTp, bp, nullptr, out, 1, B * KSEL, DO, D, 0, 0, 0);

  (void)ws_size; (void)n_in; (void)in_sizes; (void)out_size;
}
